// AdsPaiNN_27693949125345
// MI455X (gfx1250) — matrix operand order repaired and checked by reading, not compiled, not run on hardware
//
#include <hip/hip_runtime.h>
#include <hip/hip_bf16.h>
#include <math.h>

typedef __attribute__((ext_vector_type(16))) _Float16 v16h;
typedef __attribute__((ext_vector_type(8)))  _Float16 v8h;
typedef __attribute__((ext_vector_type(8)))  float    v8f;
typedef __attribute__((ext_vector_type(4)))  unsigned int v4u;
typedef __attribute__((ext_vector_type(8)))  int v8i_t;
typedef __attribute__((ext_vector_type(4)))  int v4i_t;

#define TWO_PI_F 6.28318530717958647692f
#define SSC   (1.0f/0.6f)
#define IS3C  0.5773502691896258f     // 1/sqrt(3)
#define ISHC  0.08838834764831845f    // 1/sqrt(128)
#define IS2C  0.7071067811865476f     // 1/sqrt(2)

__device__ __forceinline__ float ssilu_f(float v) {
    return v / (1.f + __expf(-v)) * SSC;
}

// ---------------- WMMA fragment loaders (16x16x32 f16, wave32) -------------
// A (16xK row-major, f16): lane m = lane&15, half h = lane>>4.
// elems 0..7  -> K = k0 + h*8 + e        (contiguous)
// elems 8..15 -> K = k0 + 16 + h*8 + e-8 (contiguous)
__device__ __forceinline__ v16h ld_a16(const _Float16* __restrict__ A, int lda,
                                       int m, int k0, int lane) {
    const _Float16* p = A + (size_t)m * lda + k0 + ((lane >> 4) << 3);
    v8h lo = *(const v8h*)p;
    v8h hi = *(const v8h*)(p + 16);
    v16h r;
#pragma unroll
    for (int e = 0; e < 8; ++e) { r[e] = lo[e]; r[e + 8] = hi[e]; }
    return r;
}

// A fragment from LDS tile with padded row stride of 136 halves (bank-spread)
__device__ __forceinline__ v16h ld_a16_sh(const _Float16* As, int k0, int lane) {
    const _Float16* p = As + (lane & 15) * 136 + k0 + ((lane >> 4) << 3);
    v8h lo = *(const v8h*)p;
    v8h hi = *(const v8h*)(p + 16);
    v16h r;
#pragma unroll
    for (int e = 0; e < 8; ++e) { r[e] = lo[e]; r[e + 8] = hi[e]; }
    return r;
}

// B loaded from pre-transposed Bt[N,K] row-major: lane n = lane&15, h = lane>>4,
// elems 0..7 -> K = k0 + h*8 + e, elems 8..15 -> K = k0 + 16 + h*8 + e-8
__device__ __forceinline__ v16h ld_b16(const _Float16* __restrict__ Bt, int ldb,
                                       int n, int k0, int lane) {
    const _Float16* p = Bt + (size_t)n * ldb + k0 + ((lane >> 4) << 3);
    v8h lo = *(const v8h*)p;
    v8h hi = *(const v8h*)(p + 16);
    v16h r;
#pragma unroll
    for (int e = 0; e < 8; ++e) { r[e] = lo[e]; r[e + 8] = hi[e]; }
    return r;
}

// ---------------- Generic tiled WMMA GEMM (16x64 tile per wave) ------------
// C[M,Nc] = act(A[M,K] @ Bt[Nc,K]^T + bias). A fragment reused across 4 B tiles.
__global__ void k_gemm(const _Float16* __restrict__ A, const _Float16* __restrict__ Bt,
                       const float* __restrict__ bias, float* __restrict__ Cf,
                       _Float16* __restrict__ Ch, int M, int Nc, int K, int act) {
    const int lane = threadIdx.x & 31;
    const int m0 = blockIdx.x << 4;
    const int n0 = blockIdx.y << 6;
    int mr = m0 + (lane & 15); if (mr >= M) mr = M - 1;
    const int nl = lane & 15;
    v8f acc0 = {}, acc1 = {}, acc2 = {}, acc3 = {};
    for (int ks = 0; ks < K; ks += 32) {
        v16h a  = ld_a16(A, K, mr, ks, lane);
        v16h b0 = ld_b16(Bt, K, n0 + nl,      ks, lane);
        v16h b1 = ld_b16(Bt, K, n0 + 16 + nl, ks, lane);
        v16h b2 = ld_b16(Bt, K, n0 + 32 + nl, ks, lane);
        v16h b3 = ld_b16(Bt, K, n0 + 48 + nl, ks, lane);
        acc0 = __builtin_amdgcn_wmma_f32_16x16x32_f16(false, a, false, b0, (short)0, acc0, false, false);
        acc1 = __builtin_amdgcn_wmma_f32_16x16x32_f16(false, a, false, b1, (short)0, acc1, false, false);
        acc2 = __builtin_amdgcn_wmma_f32_16x16x32_f16(false, a, false, b2, (short)0, acc2, false, false);
        acc3 = __builtin_amdgcn_wmma_f32_16x16x32_f16(false, a, false, b3, (short)0, acc3, false, false);
    }
    const int hh = lane >> 4;
    v8f accs[4] = { acc0, acc1, acc2, acc3 };
#pragma unroll
    for (int tb = 0; tb < 4; ++tb) {
        const int nr = n0 + (tb << 4) + nl;
        const float bv = bias ? bias[nr] : 0.f;
#pragma unroll
        for (int r = 0; r < 8; ++r) {
            int m = m0 + r + (hh << 3);
            if (m >= M) continue;
            float v = accs[tb][r] + bv;
            if (act) v = ssilu_f(v);
            if (Cf) Cf[(size_t)m * Nc + nr] = v;
            if (Ch) Ch[(size_t)m * Nc + nr] = (_Float16)v;
        }
    }
}

// ---------------- Weight convert: f32 [K,Od] -> f16 transposed [Od,Kpad] ---
__global__ void k_conv_wT(const float* __restrict__ src, _Float16* __restrict__ dst,
                          int K, int Kpad, int Od) {
    int idx = blockIdx.x * 256 + threadIdx.x;
    if (idx >= Od * Kpad) return;
    int o = idx / Kpad, k = idx - o * Kpad;
    dst[idx] = (k < K) ? (_Float16)src[(size_t)k * Od + o] : (_Float16)0.f;
}

__global__ void k_f32h(const float* __restrict__ src, _Float16* __restrict__ dst, size_t n) {
    size_t i = (size_t)blockIdx.x * blockDim.x + threadIdx.x;
    if (i < n) dst[i] = (_Float16)src[i];
}

// ---------------- Node feature build: [atom(128)|temb(128)|femb(120)|pad8] -
__global__ void k_feat(const int* __restrict__ an, const float* __restrict__ pos,
                       const float* __restrict__ t, const float* __restrict__ emb,
                       const float* __restrict__ tW, _Float16* __restrict__ feat) {
    const int n = blockIdx.x, c = threadIdx.x;   // blockDim = 384
    float v;
    if (c < 128) {
        v = emb[(size_t)an[n] * 128 + c];
    } else if (c < 256) {
        int k = c - 128;
        float ang = t[n] * TWO_PI_F * tW[k & 63];
        v = (k < 64) ? __sinf(ang) : __cosf(ang);
    } else if (c < 376) {
        int k = c - 256;
        int kk = (k < 60) ? k : k - 60;
        int d = kk / 20, f = kk - d * 20;
        float ang = pos[(size_t)n * 3 + d] * TWO_PI_F * (float)f;
        v = (k < 60) ? __sinf(ang) : __cosf(ang);
    } else {
        v = 0.f;
    }
    feat[(size_t)n * 384 + c] = (_Float16)v;
}

// ---------------- Radial basis table (f16) ---------------------------------
__global__ void k_rbf(const float* __restrict__ dist, _Float16* __restrict__ erbf, int E) {
    size_t idx = (size_t)blockIdx.x * blockDim.x + threadIdx.x;
    if (idx >= (size_t)E * 128) return;
    int e = (int)(idx >> 7);
    int r = (int)(idx & 127);
    float d = dist[e] * (1.f / 6.f);
    float diff = d - (float)r * (1.f / 127.f);
    float g = __expf(-8064.5f * diff * diff);           // -0.5/(1/127)^2
    float d2 = d * d, d4 = d2 * d2, d5 = d4 * d, d6 = d5 * d, d7 = d6 * d;
    float env = 1.f - 21.f * d5 + 35.f * d6 - 15.f * d7;
    env = (d < 1.f) ? env : 0.f;
    erbf[idx] = (_Float16)(env * g);
}

// ---------------- Degree counters ------------------------------------------
__global__ void k_count(const int* __restrict__ eidx, const unsigned char* __restrict__ fixed,
                        float* __restrict__ c1, float* __restrict__ cs, int E) {
    int e = blockIdx.x * blockDim.x + threadIdx.x;
    if (e >= E) return;
    int j = eidx[e], i = eidx[E + e];
    atomicAdd(c1 + i, 1.f);
    float w = (fixed[i] == 0 && fixed[j] == 0) ? 1.f : 0.f;
    atomicAdd(cs + i, w);
}

// ---------------- LayerNorm over H=128 (one wave per node) -----------------
__global__ void k_ln(const float* __restrict__ X, const float* __restrict__ g,
                     const float* __restrict__ b, float* __restrict__ outF,
                     _Float16* __restrict__ outH) {
    const int n = blockIdx.x, lane = threadIdx.x;
    const float* xr = X + (size_t)n * 128;
    float v[4]; float s = 0.f;
#pragma unroll
    for (int q = 0; q < 4; ++q) { v[q] = xr[lane + 32 * q]; s += v[q]; }
#pragma unroll
    for (int m = 16; m >= 1; m >>= 1) s += __shfl_xor(s, m, 32);
    float mean = s * (1.f / 128.f);
    float sq = 0.f;
#pragma unroll
    for (int q = 0; q < 4; ++q) { float d = v[q] - mean; sq += d * d; }
#pragma unroll
    for (int m = 16; m >= 1; m >>= 1) sq += __shfl_xor(sq, m, 32);
    float inv = rsqrtf(sq * (1.f / 128.f) + 1e-5f);
#pragma unroll
    for (int q = 0; q < 4; ++q) {
        int c = lane + 32 * q;
        float o = (v[q] - mean) * inv * g[c] + b[c];
        if (outF) outF[(size_t)n * 128 + c] = o;
        if (outH) outH[(size_t)n * 128 + c] = (_Float16)o;
    }
}

// ---------------- Fused edge kernel ----------------------------------------
// Block: 256 threads (8 waves), 16 edges.
// Phase 0: TDM async-loads the 16x128 f16 erbf A-tile into LDS (padded rows).
// Phase 1: rbfh[16,384] = erbf_tile @ wrT via WMMA (A from LDS) -> LDS.
// Phase 2: m = xh[j]*(rbfh+br); atomic scatter dx/dvec by destination i.
__global__ void k_edge(const _Float16* __restrict__ erbf, const _Float16* __restrict__ wrT,
                       const float* __restrict__ br, const _Float16* __restrict__ xh,
                       const float* __restrict__ vec, const int* __restrict__ eidx,
                       const float* __restrict__ rvec, const unsigned char* __restrict__ fixed,
                       int use_mask, float* __restrict__ dx, float* __restrict__ dvec, int E) {
    __shared__ float    s_rbfh[16][400];
    __shared__ _Float16 s_a[16 * 136];          // 16 rows, stride 136 halves (padded)
    __shared__ int   s_j[16], s_i[16];
    __shared__ float s_w[16];
    __shared__ float s_rv[16][3];
    const int e0 = blockIdx.x << 4;
    const int t  = threadIdx.x;
    if (t < 16) {
        int e = e0 + t; if (e >= E) e = E - 1;
        int j = eidx[e], i = eidx[E + e];
        s_j[t] = j; s_i[t] = i;
        float w = 1.f;
        if (use_mask) w = (fixed[i] == 0 && fixed[j] == 0) ? 1.f : 0.f;
        s_w[t] = w;
        __builtin_prefetch(xh + (size_t)j * 384, 0, 0);
    }
    if (t < 48) {
        int el = t / 3, d = t - el * 3;
        int e = e0 + el; if (e >= E) e = E - 1;
        s_rv[el][d] = rvec[(size_t)e * 3 + d];
    }

    int rows = E - e0; if (rows > 16) rows = 16;
#if __has_builtin(__builtin_amdgcn_tensor_load_to_lds)
    if (t < 32) {   // wave 0 issues the TDM transfer (descriptor per ISA 8.3/8.4)
        unsigned lds_base = (unsigned)(uintptr_t)(&s_a[0]);
        unsigned long long ga = (unsigned long long)(uintptr_t)(erbf + (size_t)e0 * 128);
        // group0: count=1 | lds_addr | global_addr[56:0] | type=2
        v4u g0 = { 1u, lds_base, (unsigned)ga,
                   (unsigned)((ga >> 32) & 0x01FFFFFFu) | 0x80000000u };
        // group1: data_size=2B, pad_enable, pad_interval=64dw(code5), pad_amount=4dw(code3)
        //         tensor_dim0=128, tensor_dim1=rows, tile_dim0=128, tile_dim1=rows, stride0=128
        v8i_t g1 = { (int)0x07510000u,
                     (int)(128u << 16),
                     (int)((unsigned)rows << 16),
                     (int)(128u << 16),
                     rows, 128, 0, 0 };
        v4i_t gz = { 0, 0, 0, 0 };
#if __clang_major__ >= 23
        v8i_t gz8 = { 0, 0, 0, 0, 0, 0, 0, 0 };
        __builtin_amdgcn_tensor_load_to_lds(g0, g1, gz, gz, gz8, 0);
#else
        __builtin_amdgcn_tensor_load_to_lds(g0, g1, gz, gz, 0);
#endif
        __builtin_amdgcn_s_wait_tensorcnt(0);
    }
#else
    for (int q = t; q < 16 * 128; q += 256) {
        int r = q >> 7, c = q & 127;
        int e = e0 + r; if (e >= E) e = E - 1;
        s_a[r * 136 + c] = erbf[(size_t)e * 128 + c];
    }
#endif
    __syncthreads();

    const int lane = t & 31, wave = t >> 5;
#pragma unroll
    for (int rep = 0; rep < 3; ++rep) {
        const int n0 = (wave + (rep << 3)) << 4;   // 24 n-tiles over 8 waves
        v8f acc = {};
#pragma unroll
        for (int ks = 0; ks < 128; ks += 32) {
            v16h a = ld_a16_sh(s_a, ks, lane);
            v16h b = ld_b16(wrT, 128, n0 + (lane & 15), ks, lane);
            acc = __builtin_amdgcn_wmma_f32_16x16x32_f16(false, a, false, b,
                                                         (short)0, acc, false, false);
        }
        const int col = n0 + (lane & 15);
        const int rb  = (lane >> 4) << 3;
#pragma unroll
        for (int r = 0; r < 8; ++r) s_rbfh[rb + r][col] = acc[r];
    }
    __syncthreads();

    const int el = t >> 4;
    const int cb = t & 15;
    const int e  = e0 + el;
    if (e < E) {
        const int j = s_j[el], i = s_i[el];
        const float w = s_w[el];
        if (w != 0.f) {
            const float rv0 = s_rv[el][0], rv1 = s_rv[el][1], rv2 = s_rv[el][2];
            const _Float16* xr = xh + (size_t)j * 384;
            const float* vj = vec + (size_t)j * 384;      // [3][128]
            float* dxi = dx + (size_t)i * 128;
            float* dvi = dvec + (size_t)i * 384;
#pragma unroll
            for (int q = 0; q < 8; ++q) {
                const int c = cb + (q << 4);
                float mx = (float)xr[c]       * (s_rbfh[el][c]       + br[c]);
                float m2 = (float)xr[128 + c] * (s_rbfh[el][128 + c] + br[128 + c]) * IS3C;
                float m3 = (float)xr[256 + c] * (s_rbfh[el][256 + c] + br[256 + c]);
                atomicAdd(dxi + c, mx * w);
                atomicAdd(dvi + c,       (vj[c]       * m2 + m3 * rv0) * ISHC * w);
                atomicAdd(dvi + 128 + c, (vj[128 + c] * m2 + m3 * rv1) * ISHC * w);
                atomicAdd(dvi + 256 + c, (vj[256 + c] * m2 + m3 * rv2) * ISHC * w);
            }
        }
    }
}

// ---------------- Message apply: x=(x+dx/cnt)/sqrt2, vec+=dvec/cnt ---------
__global__ void k_msg_apply(float* __restrict__ x, float* __restrict__ vec,
                            const float* __restrict__ dx, const float* __restrict__ dvec,
                            const float* __restrict__ cnt) {
    const int n = blockIdx.x, c = threadIdx.x;  // blockDim = 128
    const float ic = 1.f / fmaxf(cnt[n], 1.f);
    const size_t xo = (size_t)n * 128;
    x[xo + c] = (x[xo + c] + dx[xo + c] * ic) * IS2C;
#pragma unroll
    for (int d = 0; d < 3; ++d) {
        const size_t vo = ((size_t)n * 3 + d) * 128 + c;
        vec[vo] += dvec[vo] * ic;
    }
}

// ---------------- Update reduce: vdot, vnorm, cat=[x|vnorm] (f16) ----------
__global__ void k_upd_reduce(const float* __restrict__ V, const float* __restrict__ x,
                             float* __restrict__ vdot, _Float16* __restrict__ cat) {
    const int n = blockIdx.x, c = threadIdx.x;  // 128 threads
    float vd = 0.f, vn = 0.f;
#pragma unroll
    for (int d = 0; d < 3; ++d) {
        const float a = V[((size_t)n * 3 + d) * 256 + c];
        const float b = V[((size_t)n * 3 + d) * 256 + 128 + c];
        vd += a * b; vn += b * b;
    }
    vdot[(size_t)n * 128 + c] = vd * ISHC;
    cat[(size_t)n * 256 + c]       = (_Float16)x[(size_t)n * 128 + c];
    cat[(size_t)n * 256 + 128 + c] = (_Float16)sqrtf(vn + 1e-8f);
}

// ---------------- Update finalize ------------------------------------------
__global__ void k_upd_final(float* __restrict__ x, float* __restrict__ vec,
                            const float* __restrict__ h, const float* __restrict__ V,
                            const float* __restrict__ vdot) {
    const int n = blockIdx.x, c = threadIdx.x;  // 128 threads
    const float x1 = h[(size_t)n * 384 + c];
    const float x2 = h[(size_t)n * 384 + 128 + c];
    const float x3 = h[(size_t)n * 384 + 256 + c];
    const float dxv = (x1 + x2 * vdot[(size_t)n * 128 + c]) * IS2C;
    x[(size_t)n * 128 + c] = (x[(size_t)n * 128 + c] + dxv) * IS2C;
#pragma unroll
    for (int d = 0; d < 3; ++d)
        vec[((size_t)n * 3 + d) * 128 + c] += x3 * V[((size_t)n * 3 + d) * 256 + c];
}

// ---------------- GEB: vec1 = ||v@wv1|| over axis 1; cat=[x|vec1] ----------
__global__ void k_geb_reduce(const float* __restrict__ W1v, const float* __restrict__ x,
                             _Float16* __restrict__ cat, int hi) {
    const int n = blockIdx.x, c = threadIdx.x;  // blockDim = hi
    float s = 0.f;
#pragma unroll
    for (int d = 0; d < 3; ++d) {
        const float v = W1v[((size_t)n * 3 + d) * hi + c];
        s += v * v;
    }
    cat[(size_t)n * 2 * hi + c]      = (_Float16)x[(size_t)n * hi + c];
    cat[(size_t)n * 2 * hi + hi + c] = (_Float16)sqrtf(s);
}

__global__ void k_geb_final(const float* __restrict__ h, const float* __restrict__ vec2,
                            float* __restrict__ xo_out, float* __restrict__ vec_out, int ho) {
    const int n = blockIdx.x, c = threadIdx.x;  // blockDim = ho
    const float xo = h[(size_t)n * 2 * ho + c];
    const float vg = h[(size_t)n * 2 * ho + ho + c];
    xo_out[(size_t)n * ho + c] = ssilu_f(xo);
#pragma unroll
    for (int d = 0; d < 3; ++d)
        vec_out[((size_t)n * 3 + d) * ho + c] = vg * vec2[((size_t)n * 3 + d) * ho + c];
}

// ---------------- GEB2 small ops (ho = 1) ----------------------------------
__global__ void k_geb2_vec2(const float* __restrict__ vecg, const float* __restrict__ wv2,
                            float* __restrict__ vec2b, int n3) {
    int idx = blockIdx.x * blockDim.x + threadIdx.x;
    if (idx >= n3) return;
    const float* v = vecg + (size_t)idx * 64;
    float s = 0.f;
#pragma unroll
    for (int k = 0; k < 64; ++k) s += v[k] * wv2[k];
    vec2b[idx] = s;
}

__global__ void k_geb2_final(const _Float16* __restrict__ g1, const float* __restrict__ w2,
                             const float* __restrict__ b2, const float* __restrict__ vec2b,
                             float* __restrict__ out, int N) {
    int n = blockIdx.x * blockDim.x + threadIdx.x;
    if (n >= N) return;
    const _Float16* g = g1 + (size_t)n * 64;
    float vg = b2[1];
#pragma unroll
    for (int k = 0; k < 64; ++k) vg += (float)g[k] * w2[k * 2 + 1];
#pragma unroll
    for (int d = 0; d < 3; ++d)
        out[(size_t)n * 3 + d] = vg * vec2b[(size_t)n * 3 + d];
}

// ===========================================================================
extern "C" void kernel_launch(void* const* d_in, const int* in_sizes, int n_in,
                              void* d_out, int out_size, void* d_ws, size_t ws_size,
                              hipStream_t stream) {
    const int N = in_sizes[0];
    const int E = in_sizes[4];
    const int N3 = 3 * N;

    const int*   an    = (const int*)d_in[0];
    const float* pos   = (const float*)d_in[1];
    const float* tt    = (const float*)d_in[2];
    const int*   eidx  = (const int*)d_in[3];
    const float* dist  = (const float*)d_in[4];
    const float* rvec  = (const float*)d_in[5];
    const unsigned char* fixed = (const unsigned char*)d_in[6];

    int p = 7;
    auto F = [&](int i) { return (const float*)d_in[i]; };
    const float* emb_table = F(p++);
    const float* tWv       = F(p++);
    const float* Wemb      = F(p++);
    const float* bemb      = F(p++);
    struct MsgP { const float *ln_g, *ln_b, *w1, *b1, *w2, *b2, *wr, *br; };
    struct UpdP { const float *wv, *w1, *b1, *w2, *b2; };
    struct GebP { const float *wv1, *wv2, *w1, *b1, *w2, *b2; };
    auto rdMsg = [&]() { MsgP m; m.ln_g=F(p++); m.ln_b=F(p++); m.w1=F(p++); m.b1=F(p++);
                         m.w2=F(p++); m.b2=F(p++); m.wr=F(p++); m.br=F(p++); return m; };
    auto rdUpd = [&]() { UpdP u; u.wv=F(p++); u.w1=F(p++); u.b1=F(p++); u.w2=F(p++);
                         u.b2=F(p++); return u; };
    auto rdGeb = [&]() { GebP g; g.wv1=F(p++); g.wv2=F(p++); g.w1=F(p++); g.b1=F(p++);
                         g.w2=F(p++); g.b2=F(p++); return g; };
    MsgP msgs[7]; UpdP upds[7];
    msgs[0] = rdMsg();
    const float* surf_ln_g = F(p++);
    const float* surf_ln_b = F(p++);
    upds[0] = rdUpd();
    GebP g1p = rdGeb();
    GebP g2p = rdGeb();
    for (int l = 1; l < 7; ++l) msgs[l] = rdMsg();
    for (int l = 1; l < 7; ++l) upds[l] = rdUpd();

    // ---- workspace carve-up ----
    char* wsb = (char*)d_ws;
    size_t off = 0;
    auto alloc = [&](size_t bytes) -> void* {
        void* r = wsb + off;
        off = (off + bytes + 255) & ~(size_t)255;
        return r;
    };
    _Float16* feat  = (_Float16*)alloc((size_t)N * 384 * 2);
    float*    x     = (float*)   alloc((size_t)N * 128 * 4);
    _Float16* xln   = (_Float16*)alloc((size_t)N * 128 * 2);
    _Float16* t1h   = (_Float16*)alloc((size_t)N * 128 * 2);
    _Float16* xhh   = (_Float16*)alloc((size_t)N * 384 * 2);
    float*    vec   = (float*)   alloc((size_t)N3 * 128 * 4);
    _Float16* vech  = (_Float16*)alloc((size_t)N3 * 128 * 2);
    _Float16* erbfh = (_Float16*)alloc((size_t)E * 128 * 2);
    float*    dx    = (float*)   alloc((size_t)N * 128 * 4);
    float*    dvec  = (float*)   alloc((size_t)N3 * 128 * 4);
    float*    cnt1  = (float*)   alloc((size_t)N * 4);
    float*    cnts  = (float*)   alloc((size_t)N * 4);
    float*    V     = (float*)   alloc((size_t)N3 * 256 * 4);
    float*    vdot  = (float*)   alloc((size_t)N * 128 * 4);
    _Float16* cath  = (_Float16*)alloc((size_t)N * 256 * 2);
    float*    hbuf  = (float*)   alloc((size_t)N * 384 * 4);
    float*    vec2a = (float*)   alloc((size_t)N3 * 64 * 4);
    float*    xg    = (float*)   alloc((size_t)N * 64 * 4);
    float*    vecg  = (float*)   alloc((size_t)N3 * 64 * 4);
    _Float16* vecgh = (_Float16*)alloc((size_t)N3 * 64 * 2);
    float*    vec2b = (float*)   alloc((size_t)N3 * 4);

    auto convW = [&](const float* src, int K, int Kpad, int Od) -> const _Float16* {
        _Float16* dst = (_Float16*)alloc((size_t)Od * Kpad * 2);
        int tot = Od * Kpad;
        hipLaunchKernelGGL(k_conv_wT, dim3((tot + 255) / 256), dim3(256), 0, stream,
                           src, dst, K, Kpad, Od);
        return dst;
    };

    const _Float16* WembT = convW(Wemb, 376, 384, 128);
    const _Float16 *mw1[7], *mw2[7], *mwr[7], *uwv[7], *uw1[7], *uw2[7];
    for (int l = 0; l < 7; ++l) {
        mw1[l] = convW(msgs[l].w1, 128, 128, 128);
        mw2[l] = convW(msgs[l].w2, 128, 128, 384);
        mwr[l] = convW(msgs[l].wr, 128, 128, 384);
        uwv[l] = convW(upds[l].wv, 128, 128, 256);
        uw1[l] = convW(upds[l].w1, 256, 256, 128);
        uw2[l] = convW(upds[l].w2, 128, 128, 384);
    }
    const _Float16* g1wv1T = convW(g1p.wv1, 128, 128, 128);
    const _Float16* g1wv2T = convW(g1p.wv2, 128, 128, 64);
    const _Float16* g1w1T  = convW(g1p.w1,  256, 256, 128);
    const _Float16* g1w2T  = convW(g1p.w2,  128, 128, 128);
    const _Float16* g2wv1T = convW(g2p.wv1, 64, 64, 64);
    const _Float16* g2w1T  = convW(g2p.w1,  128, 128, 64);

    auto gemm = [&](const _Float16* A, const _Float16* Bt, const float* bias,
                    float* Cf, _Float16* Ch, int M, int Nc, int K, int act) {
        hipLaunchKernelGGL(k_gemm, dim3((M + 15) / 16, Nc / 64), dim3(32), 0, stream,
                           A, Bt, bias, Cf, Ch, M, Nc, K, act);
    };

    // ---- prologue ----
    hipMemsetAsync(cnt1, 0, (size_t)N * 4, stream);
    hipMemsetAsync(cnts, 0, (size_t)N * 4, stream);
    hipMemsetAsync(vec, 0, (size_t)N3 * 128 * 4, stream);
    hipLaunchKernelGGL(k_count, dim3((E + 255) / 256), dim3(256), 0, stream,
                       eidx, fixed, cnt1, cnts, E);
    hipLaunchKernelGGL(k_feat, dim3(N), dim3(384), 0, stream,
                       an, pos, tt, emb_table, tWv, feat);
    {
        size_t tot = (size_t)E * 128;
        hipLaunchKernelGGL(k_rbf, dim3((unsigned)((tot + 255) / 256)), dim3(256), 0, stream,
                           dist, erbfh, E);
    }
    gemm(feat, WembT, bemb, x, nullptr, N, 128, 384, 0);

    auto message = [&](int li, int use_mask, const float* cnt) {
        const MsgP& mp = msgs[li];
        hipLaunchKernelGGL(k_ln, dim3(N), dim3(32), 0, stream,
                           x, mp.ln_g, mp.ln_b, (float*)nullptr, xln);
        gemm(xln, mw1[li], mp.b1, nullptr, t1h, N, 128, 128, 1);
        gemm(t1h, mw2[li], mp.b2, nullptr, xhh, N, 384, 128, 0);
        hipMemsetAsync(dx, 0, (size_t)N * 128 * 4, stream);
        hipMemsetAsync(dvec, 0, (size_t)N3 * 128 * 4, stream);
        hipLaunchKernelGGL(k_edge, dim3((E + 15) / 16), dim3(256), 0, stream,
                           erbfh, mwr[li], mp.br, xhh, vec, eidx, rvec, fixed,
                           use_mask, dx, dvec, E);
        hipLaunchKernelGGL(k_msg_apply, dim3(N), dim3(128), 0, stream,
                           x, vec, dx, dvec, cnt);
    };
    auto update = [&](int li) {
        const UpdP& up = upds[li];
        hipLaunchKernelGGL(k_f32h, dim3((unsigned)(((size_t)N3 * 128 + 255) / 256)),
                           dim3(256), 0, stream, vec, vech, (size_t)N3 * 128);
        gemm(vech, uwv[li], nullptr, V, nullptr, N3, 256, 128, 0);
        hipLaunchKernelGGL(k_upd_reduce, dim3(N), dim3(128), 0, stream, V, x, vdot, cath);
        gemm(cath, uw1[li], up.b1, nullptr, t1h, N, 128, 256, 1);
        gemm(t1h, uw2[li], up.b2, hbuf, nullptr, N, 384, 128, 0);
        hipLaunchKernelGGL(k_upd_final, dim3(N), dim3(128), 0, stream, x, vec, hbuf, V, vdot);
    };

    // ---- surface interaction ----
    message(0, 1, cnts);
    hipLaunchKernelGGL(k_ln, dim3(N), dim3(32), 0, stream,
                       x, surf_ln_g, surf_ln_b, x, (_Float16*)nullptr);
    update(0);

    // ---- main layers ----
    for (int l = 1; l < 7; ++l) {
        message(l, 0, cnt1);
        update(l);
    }

    // ---- GEB 1 (hi=128 -> ho=64) ----
    hipLaunchKernelGGL(k_f32h, dim3((unsigned)(((size_t)N3 * 128 + 255) / 256)),
                       dim3(256), 0, stream, vec, vech, (size_t)N3 * 128);
    gemm(vech, g1wv1T, nullptr, V, nullptr, N3, 128, 128, 0);      // W1v
    gemm(vech, g1wv2T, nullptr, vec2a, nullptr, N3, 64, 128, 0);   // vec2
    hipLaunchKernelGGL(k_geb_reduce, dim3(N), dim3(128), 0, stream, V, x, cath, 128);
    gemm(cath, g1w1T, g1p.b1, nullptr, t1h, N, 128, 256, 1);
    gemm(t1h, g1w2T, g1p.b2, hbuf, nullptr, N, 128, 128, 0);
    hipLaunchKernelGGL(k_geb_final, dim3(N), dim3(64), 0, stream, hbuf, vec2a, xg, vecg, 64);

    // ---- GEB 2 (hi=64 -> ho=1) ----
    hipLaunchKernelGGL(k_f32h, dim3((unsigned)(((size_t)N3 * 64 + 255) / 256)),
                       dim3(256), 0, stream, vecg, vecgh, (size_t)N3 * 64);
    gemm(vecgh, g2wv1T, nullptr, V, nullptr, N3, 64, 64, 0);       // W1v (reuse V)
    hipLaunchKernelGGL(k_geb2_vec2, dim3((N3 + 255) / 256), dim3(256), 0, stream,
                       vecg, g2p.wv2, vec2b, N3);
    hipLaunchKernelGGL(k_geb_reduce, dim3(N), dim3(64), 0, stream, V, xg, cath, 64);
    gemm(cath, g2w1T, g2p.b1, nullptr, t1h, N, 64, 128, 1);
    hipLaunchKernelGGL(k_geb2_final, dim3((N + 63) / 64), dim3(64), 0, stream,
                       t1h, g2p.w2, g2p.b2, vec2b, (float*)d_out, N);
}
